// AttnBlock_35175782154905
// MI455X (gfx1250) — matrix operand order repaired and checked by reading, not compiled, not run on hardware
//
#include <hip/hip_runtime.h>
#include <hip/hip_bf16.h>
#include <stddef.h>

typedef __attribute__((ext_vector_type(16))) _Float16 v16h;
typedef __attribute__((ext_vector_type(8)))  float    v8f;

#define BATCH     16
#define CCH       512
#define HWPIX     1024
#define NGRP      16
#define GRP_ELEMS ((CCH / NGRP) * HWPIX)   /* 32768, contiguous per (b,g) */
#define GN_EPS    1e-6f
#define SM_SCALE  0.04419417382415922f     /* 1/sqrt(512) */

// ---------------------------------------------------------------------------
// WMMA helpers (CDNA5 gfx1250, wave32). Layouts per ISA 7.12.2.
// ---------------------------------------------------------------------------
__device__ __forceinline__ v8f wmma_f16(v16h a, v16h b, v8f c) {
  return __builtin_amdgcn_wmma_f32_16x16x32_f16(false, a, false, b, (short)0, c,
                                                false, false);
}

// A fragment: 16x32 (MxK) f16 from row-major src[ld halves per row].
// lane m = lane&15 ; K = (j&7) + 8*(lane>>4) + 16*(j>>3)
// -> per lane two contiguous 8-half (16 B) runs => 2x b128 loads.
__device__ __forceinline__ v16h load_a_frag(const _Float16* src, int ld) {
  const int lane = threadIdx.x & 31;
  const int g = lane >> 4, m = lane & 15;
  v16h a;
#pragma unroll
  for (int j = 0; j < 16; ++j) {
    const int k = (j & 7) + 8 * g + 16 * (j >> 3);
    a[j] = src[m * ld + k];
  }
  return a;
}

// B fragment: 32x16 (KxN) f16 sourced from an N-major buffer srcT[N][K]
// (ldT halves per N-row). lane n = lane&15 ; K = (j&7) + 8*(lane>>4) + 16*(j>>3)
// -> per lane two contiguous 8-half (16 B) runs => 2x b128 loads.
__device__ __forceinline__ v16h load_bT_frag(const _Float16* srcT, int ldT) {
  const int lane = threadIdx.x & 31;
  const int g = lane >> 4, n = lane & 15;
  v16h b;
#pragma unroll
  for (int j = 0; j < 16; ++j) b[j] = srcT[n * ldT + (j & 7) + 8 * g + 16 * (j >> 3)];
  return b;
}

// Async global->LDS 16B copy (CDNA5 ASYNCcnt path, ISA 10.x / 15.18.3).
__device__ __forceinline__ void async_copy_b128(const _Float16* gsrc,
                                                _Float16* ldst) {
  unsigned lds_off = (unsigned)(size_t)ldst;  // low 32 bits = LDS byte offset
  asm volatile("global_load_async_to_lds_b128 %0, %1, off" ::"v"(lds_off),
               "v"(gsrc)
               : "memory");
}
__device__ __forceinline__ void async_wait0() {
  asm volatile("s_wait_asynccnt 0x0" ::: "memory");
}

// ---------------------------------------------------------------------------
// Kernel 1: f32 -> f16 weight conversion
// ---------------------------------------------------------------------------
__global__ void cvt_f16_kernel(const float* __restrict__ src,
                               _Float16* __restrict__ dst, int n) {
  int i = blockIdx.x * blockDim.x + threadIdx.x;
  if (i < n) dst[i] = (_Float16)src[i];
}

// ---------------------------------------------------------------------------
// Kernel 2: GroupNorm statistics. One block per (b,g); group data contiguous.
// ---------------------------------------------------------------------------
__global__ void gn_stats_kernel(const float* __restrict__ x,
                                float* __restrict__ meanOut,
                                float* __restrict__ rstdOut) {
  __shared__ float sSum[256];
  __shared__ float sSq[256];
  const int bg = blockIdx.x;
  const float4* base = (const float4*)(x + (size_t)bg * GRP_ELEMS);
  const int t = threadIdx.x;
  float s = 0.f, sq = 0.f;
  for (int i = t; i < GRP_ELEMS / 4; i += 256) {
    float4 v = base[i];
    s += v.x + v.y + v.z + v.w;
    sq += v.x * v.x + v.y * v.y + v.z * v.z + v.w * v.w;
  }
  sSum[t] = s;
  sSq[t] = sq;
  __syncthreads();
  for (int st = 128; st > 0; st >>= 1) {
    if (t < st) {
      sSum[t] += sSum[t + st];
      sSq[t] += sSq[t + st];
    }
    __syncthreads();
  }
  if (t == 0) {
    float m = sSum[0] * (1.0f / GRP_ELEMS);
    float var = sSq[0] * (1.0f / GRP_ELEMS) - m * m;
    var = var < 0.f ? 0.f : var;
    meanOut[bg] = m;
    rstdOut[bg] = rsqrtf(var + GN_EPS);
  }
}

// ---------------------------------------------------------------------------
// Kernel 3: QKV projection GEMM with fused GroupNorm on the B operand.
// out[oc,p] = sum_c W[oc,c] * hnorm[c,p] + bias[oc]   (per batch)
// Block tile 128(M) x 64(N); 8 waves of 32x32; K loop 512 in steps of 32.
// TRANSPOSED=1 -> dst[b][p][oc] (Q,K) ; TRANSPOSED=0 -> dst[b][oc][p] (V).
// ---------------------------------------------------------------------------
template <int TRANSPOSED>
__global__ void qkv_gemm_kernel(const _Float16* __restrict__ w16,  // [C][C]
                                const float* __restrict__ bias,    // [C]
                                const float* __restrict__ x,       // [B][C][HW]
                                const float* __restrict__ gamma,
                                const float* __restrict__ beta,
                                const float* __restrict__ mean,    // [B*G]
                                const float* __restrict__ rstd,    // [B*G]
                                _Float16* __restrict__ dst) {
  __shared__ alignas(16) _Float16 ldsA[128 * 32];   // [M][K]
  __shared__ alignas(16) _Float16 ldsBT[64 * 32];   // [N][K] (transposed)
  const int bat = blockIdx.z;
  const int mBase = blockIdx.y * 128;
  const int pBase = blockIdx.x * 64;
  const int t = threadIdx.x;
  const int wave = t >> 5, lane = t & 31;
  const int g = lane >> 4, ln = lane & 15;
  const int wm = wave >> 1, wn = wave & 1;  // wave tile (wm*32, wn*32)
  const float* xb = x + (size_t)bat * CCH * HWPIX;

  v8f acc[2][2];
#pragma unroll
  for (int mi = 0; mi < 2; ++mi)
#pragma unroll
    for (int ni = 0; ni < 2; ++ni) acc[mi][ni] = (v8f){};

  for (int k0 = 0; k0 < CCH; k0 += 32) {
    if (k0 + 32 < CCH)
      __builtin_prefetch(&w16[(size_t)mBase * CCH + k0 + 32], 0, 0);
    // Stage A: W[mBase..+127][k0..+31], 16B vector copies
#pragma unroll
    for (int u = 0; u < 2; ++u) {
      int v = t * 2 + u;          // 0..511
      int r = v >> 2, seg = v & 3;
      *(uint4*)&ldsA[r * 32 + seg * 8] =
          *(const uint4*)&w16[(size_t)(mBase + r) * CCH + k0 + seg * 8];
    }
    // Stage B (transposed): ldsBT[p][c] = groupnorm(x)[k0+c][pBase+p]
#pragma unroll
    for (int u = 0; u < 2; ++u) {
      int v = t * 2 + u;          // 0..511
      int r = v >> 4, quad = v & 15;
      int c = k0 + r;
      int grp = bat * NGRP + (c >> 5);
      float mu = mean[grp], rs = rstd[grp];
      float ga = gamma[c], be = beta[c];
      float4 xv = *(const float4*)&xb[(size_t)c * HWPIX + pBase + quad * 4];
      ldsBT[(quad * 4 + 0) * 32 + r] = (_Float16)((xv.x - mu) * rs * ga + be);
      ldsBT[(quad * 4 + 1) * 32 + r] = (_Float16)((xv.y - mu) * rs * ga + be);
      ldsBT[(quad * 4 + 2) * 32 + r] = (_Float16)((xv.z - mu) * rs * ga + be);
      ldsBT[(quad * 4 + 3) * 32 + r] = (_Float16)((xv.w - mu) * rs * ga + be);
    }
    __syncthreads();
    v16h afr[2], bfr[2];
#pragma unroll
    for (int mi = 0; mi < 2; ++mi)
      afr[mi] = load_a_frag(&ldsA[(wm * 32 + mi * 16) * 32], 32);
#pragma unroll
    for (int ni = 0; ni < 2; ++ni)
      bfr[ni] = load_bT_frag(&ldsBT[(wn * 32 + ni * 16) * 32], 32);
#pragma unroll
    for (int mi = 0; mi < 2; ++mi)
#pragma unroll
      for (int ni = 0; ni < 2; ++ni)
        acc[mi][ni] = wmma_f16(afr[mi], bfr[ni], acc[mi][ni]);
    __syncthreads();
  }

#pragma unroll
  for (int mi = 0; mi < 2; ++mi)
#pragma unroll
    for (int ni = 0; ni < 2; ++ni)
#pragma unroll
      for (int r = 0; r < 8; ++r) {
        int oM = mBase + wm * 32 + mi * 16 + r + 8 * g;
        int oN = pBase + wn * 32 + ni * 16 + ln;
        float val = acc[mi][ni][r] + bias[oM];
        if constexpr (TRANSPOSED)
          dst[((size_t)bat * HWPIX + oN) * CCH + oM] = (_Float16)val;
        else
          dst[((size_t)bat * CCH + oM) * HWPIX + oN] = (_Float16)val;
      }
}

// ---------------------------------------------------------------------------
// Kernel 4: flash attention per (batch, 32-query tile).
// qT [B][HW][C], kT [B][HW][C], vM [B][C][HW] f16 -> oT [B][HW][C] f16.
// All global B-fragments are two 16 B runs per lane (N-major layouts).
// ---------------------------------------------------------------------------
__global__ void attn_kernel(const _Float16* __restrict__ qT,
                            const _Float16* __restrict__ kT,
                            const _Float16* __restrict__ vM,
                            _Float16* __restrict__ oT) {
  __shared__ alignas(16) _Float16 ldsQ[32 * CCH];  // 32 KB, [q][c]
  __shared__ alignas(16) float    ldsS[32 * 64];   // 8 KB
  __shared__ alignas(16) _Float16 ldsP[32 * 64];   // 4 KB, [q][k]
  __shared__ float rowM[32], rowL[32], rowAlpha[32];

  const int bat = blockIdx.y;
  const int qBase = blockIdx.x * 32;
  const int t = threadIdx.x;
  const int wave = t >> 5, lane = t & 31;
  const int g = lane >> 4, ln = lane & 15;

  // Load Q tile (rows qBase..+31 contiguous in qT), 16B vector copies
  const _Float16* qTb = qT + ((size_t)bat * HWPIX + qBase) * CCH;
#pragma unroll
  for (int i = 0; i < 8; ++i) {
    int v = t + i * 256;  // 0..2047 uint4s
    ((uint4*)ldsQ)[v] = ((const uint4*)qTb)[v];
  }
  if (t < 32) {
    rowM[t] = -1e30f;
    rowL[t] = 0.f;
    rowAlpha[t] = 0.f;
  }
  __syncthreads();

  const int smi = wave >> 2, sni = wave & 3;  // S-frag coords
  const int chanBase = wave * 64;             // O channel slice
  const _Float16* kTb = kT + (size_t)bat * HWPIX * CCH;
  const _Float16* vMb = vM + (size_t)bat * CCH * HWPIX;

  v8f oacc[2][4];
#pragma unroll
  for (int mi = 0; mi < 2; ++mi)
#pragma unroll
    for (int ni = 0; ni < 4; ++ni) oacc[mi][ni] = (v8f){};

  for (int kb = 0; kb < HWPIX; kb += 64) {
    // ---- S = Q^T K : each wave one 16x16 frag, K-loop over channels ----
    v8f s = (v8f){};
    const _Float16* kRow = kTb + (size_t)(kb + sni * 16) * CCH;
#pragma unroll 4
    for (int c0 = 0; c0 < CCH; c0 += 32) {
      v16h a = load_a_frag(&ldsQ[(smi * 16) * CCH + c0], CCH);
      v16h b = load_bT_frag(kRow + c0, CCH);
      s = wmma_f16(a, b, s);
    }
#pragma unroll
    for (int r = 0; r < 8; ++r)
      ldsS[(smi * 16 + r + 8 * g) * 64 + sni * 16 + ln] = s[r];
    if (kb + 64 < HWPIX)
      __builtin_prefetch(kTb + (size_t)(kb + 64) * CCH, 0, 0);
    __syncthreads();

    // ---- online softmax over this 64-key block (8 threads per q row) ----
    {
      const int row = t >> 3, sub = t & 7;
      float vals[8];
      float mloc = -1e30f;
#pragma unroll
      for (int i = 0; i < 8; ++i) {
        vals[i] = ldsS[row * 64 + sub * 8 + i] * SM_SCALE;
        mloc = fmaxf(mloc, vals[i]);
      }
#pragma unroll
      for (int off = 4; off; off >>= 1)
        mloc = fmaxf(mloc, __shfl_xor(mloc, off, 8));
      const float mprev = rowM[row];
      const float mnew = fmaxf(mprev, mloc);
      const float alpha = __expf(mprev - mnew);
      float psum = 0.f;
#pragma unroll
      for (int i = 0; i < 8; ++i) {
        float p = __expf(vals[i] - mnew);
        ldsP[row * 64 + sub * 8 + i] = (_Float16)p;
        psum += p;
      }
#pragma unroll
      for (int off = 4; off; off >>= 1) psum += __shfl_xor(psum, off, 8);
      if (sub == 0) {
        rowM[row] = mnew;
        rowL[row] = rowL[row] * alpha + psum;
        rowAlpha[row] = alpha;
      }
    }
    __syncthreads();

    // ---- O = O*alpha + P V^T (each wave: its 64-channel slice) ----
#pragma unroll
    for (int mi = 0; mi < 2; ++mi) {
#pragma unroll
      for (int r = 0; r < 8; ++r) {
        const float al = rowAlpha[mi * 16 + r + 8 * g];
#pragma unroll
        for (int ni = 0; ni < 4; ++ni) oacc[mi][ni][r] *= al;
      }
    }
#pragma unroll
    for (int kk = 0; kk < 2; ++kk) {
      v16h pa0 = load_a_frag(&ldsP[0 * 64 + kk * 32], 64);
      v16h pa1 = load_a_frag(&ldsP[16 * 64 + kk * 32], 64);
#pragma unroll
      for (int ni = 0; ni < 4; ++ni) {
        // B = V^T[key][ch]; vM is [ch][key] i.e. N-major for this product.
        v16h vb = load_bT_frag(
            vMb + (size_t)(chanBase + ni * 16) * HWPIX + kb + kk * 32, HWPIX);
        oacc[0][ni] = wmma_f16(pa0, vb, oacc[0][ni]);
        oacc[1][ni] = wmma_f16(pa1, vb, oacc[1][ni]);
      }
    }
    __syncthreads();
  }

  // ---- finalize: divide by l, write oT[b][q][c] ----
#pragma unroll
  for (int mi = 0; mi < 2; ++mi)
#pragma unroll
    for (int ni = 0; ni < 4; ++ni)
#pragma unroll
      for (int r = 0; r < 8; ++r) {
        const int m = mi * 16 + r + 8 * g;
        const int ch = chanBase + ni * 16 + ln;
        float val = oacc[mi][ni][r] / rowL[m];
        oT[((size_t)bat * HWPIX + qBase + m) * CCH + ch] = (_Float16)val;
      }
}

// ---------------------------------------------------------------------------
// Kernel 5: output projection + bias + residual, f32 out.
// B tile staged into LDS via async global->LDS b128 copies (ASYNCcnt path).
// ---------------------------------------------------------------------------
__global__ void out_proj_kernel(const _Float16* __restrict__ wp16,  // [C][C]
                                const float* __restrict__ bp,       // [C]
                                const _Float16* __restrict__ oT,    // [B][HW][C]
                                const float* __restrict__ x,        // [B][C][HW]
                                float* __restrict__ out) {
  __shared__ alignas(16) _Float16 ldsA[128 * 32];   // [M][K]
  __shared__ alignas(16) _Float16 ldsBT[64 * 32];   // [N][K]
  const int bat = blockIdx.z;
  const int mBase = blockIdx.y * 128;
  const int pBase = blockIdx.x * 64;
  const int t = threadIdx.x;
  const int wave = t >> 5, lane = t & 31;
  const int g = lane >> 4, ln = lane & 15;
  const int wm = wave >> 1, wn = wave & 1;
  const _Float16* oTb = oT + (size_t)bat * HWPIX * CCH;

  v8f acc[2][2];
#pragma unroll
  for (int mi = 0; mi < 2; ++mi)
#pragma unroll
    for (int ni = 0; ni < 2; ++ni) acc[mi][ni] = (v8f){};

  for (int k0 = 0; k0 < CCH; k0 += 32) {
    if (k0 + 32 < CCH)
      __builtin_prefetch(&wp16[(size_t)mBase * CCH + k0 + 32], 0, 0);
    // Stage A: Wp tile, 16B vector copies
#pragma unroll
    for (int u = 0; u < 2; ++u) {
      int v = t * 2 + u;
      int r = v >> 2, seg = v & 3;
      *(uint4*)&ldsA[r * 32 + seg * 8] =
          *(const uint4*)&wp16[(size_t)(mBase + r) * CCH + k0 + seg * 8];
    }
    // Stage B^T: ldsBT[p][c] <- oT[pBase+p][k0+c], one async b128 per thread
    {
      int row = t >> 2, seg = t & 3;  // 64 rows x 4 x 16B
      async_copy_b128(&oTb[(size_t)(pBase + row) * CCH + k0 + seg * 8],
                      &ldsBT[row * 32 + seg * 8]);
      async_wait0();
    }
    __syncthreads();
    v16h afr[2], bfr[2];
#pragma unroll
    for (int mi = 0; mi < 2; ++mi)
      afr[mi] = load_a_frag(&ldsA[(wm * 32 + mi * 16) * 32], 32);
#pragma unroll
    for (int ni = 0; ni < 2; ++ni)
      bfr[ni] = load_bT_frag(&ldsBT[(wn * 32 + ni * 16) * 32], 32);
#pragma unroll
    for (int mi = 0; mi < 2; ++mi)
#pragma unroll
      for (int ni = 0; ni < 2; ++ni)
        acc[mi][ni] = wmma_f16(afr[mi], bfr[ni], acc[mi][ni]);
    __syncthreads();
  }

#pragma unroll
  for (int mi = 0; mi < 2; ++mi)
#pragma unroll
    for (int ni = 0; ni < 2; ++ni)
#pragma unroll
      for (int r = 0; r < 8; ++r) {
        int oR = mBase + wm * 32 + mi * 16 + r + 8 * g;
        int oN = pBase + wn * 32 + ni * 16 + ln;
        size_t idx = ((size_t)bat * CCH + oR) * HWPIX + oN;
        out[idx] = acc[mi][ni][r] + bp[oR] + x[idx];
      }
}

// ---------------------------------------------------------------------------
// Launch
// ---------------------------------------------------------------------------
extern "C" void kernel_launch(void* const* d_in, const int* in_sizes, int n_in,
                              void* d_out, int out_size, void* d_ws,
                              size_t ws_size, hipStream_t stream) {
  (void)in_sizes; (void)n_in; (void)out_size; (void)ws_size;
  const float* x     = (const float*)d_in[0];
  const float* gamma = (const float*)d_in[1];
  const float* beta  = (const float*)d_in[2];
  const float* Wq    = (const float*)d_in[3];
  const float* bq    = (const float*)d_in[4];
  const float* Wk    = (const float*)d_in[5];
  const float* bk    = (const float*)d_in[6];
  const float* Wv    = (const float*)d_in[7];
  const float* bv    = (const float*)d_in[8];
  const float* Wp    = (const float*)d_in[9];
  const float* bp    = (const float*)d_in[10];
  float* out = (float*)d_out;

  char* ws = (char*)d_ws;
  size_t off = 0;
  const size_t wBytes = (size_t)CCH * CCH * sizeof(_Float16);             // 512 KB
  const size_t actBytes = (size_t)BATCH * CCH * HWPIX * sizeof(_Float16); // 16 MB
  _Float16* wq16 = (_Float16*)(ws + off); off += wBytes;
  _Float16* wk16 = (_Float16*)(ws + off); off += wBytes;
  _Float16* wv16 = (_Float16*)(ws + off); off += wBytes;
  _Float16* wp16 = (_Float16*)(ws + off); off += wBytes;
  float* gnMean  = (float*)(ws + off); off += BATCH * NGRP * sizeof(float);
  float* gnRstd  = (float*)(ws + off); off += BATCH * NGRP * sizeof(float);
  _Float16* qT = (_Float16*)(ws + off); off += actBytes;  // [B][HW][C]
  _Float16* kT = (_Float16*)(ws + off); off += actBytes;  // [B][HW][C]
  _Float16* vM = (_Float16*)(ws + off); off += actBytes;  // [B][C][HW]
  _Float16* oT = (_Float16*)(ws + off); off += actBytes;  // [B][HW][C]

  // 1) weights f32 -> f16
  const int wN = CCH * CCH;
  cvt_f16_kernel<<<wN / 256, 256, 0, stream>>>(Wq, wq16, wN);
  cvt_f16_kernel<<<wN / 256, 256, 0, stream>>>(Wk, wk16, wN);
  cvt_f16_kernel<<<wN / 256, 256, 0, stream>>>(Wv, wv16, wN);
  cvt_f16_kernel<<<wN / 256, 256, 0, stream>>>(Wp, wp16, wN);

  // 2) groupnorm stats
  gn_stats_kernel<<<BATCH * NGRP, 256, 0, stream>>>(x, gnMean, gnRstd);

  // 3) fused GN + QKV projections (WMMA)
  dim3 gGemm(HWPIX / 64, CCH / 128, BATCH);
  qkv_gemm_kernel<1><<<gGemm, 256, 0, stream>>>(wq16, bq, x, gamma, beta,
                                                gnMean, gnRstd, qT);
  qkv_gemm_kernel<1><<<gGemm, 256, 0, stream>>>(wk16, bk, x, gamma, beta,
                                                gnMean, gnRstd, kT);
  qkv_gemm_kernel<0><<<gGemm, 256, 0, stream>>>(wv16, bv, x, gamma, beta,
                                                gnMean, gnRstd, vM);

  // 4) flash attention (WMMA both GEMMs, online softmax)
  dim3 gAttn(HWPIX / 32, BATCH);
  attn_kernel<<<gAttn, 256, 0, stream>>>(qT, kT, vM, oT);

  // 5) output projection + residual (WMMA, async B staging)
  out_proj_kernel<<<gGemm, 256, 0, stream>>>(wp16, bp, oT, x, out);
}
